// NawrotDownsampler_75453985456505
// MI455X (gfx1250) — matrix operand order repaired and checked by reading, not compiled, not run on hardware
//
#include <hip/hip_runtime.h>
#include <hip/hip_bf16.h>

typedef __attribute__((ext_vector_type(16))) _Float16 v16h;
typedef __attribute__((ext_vector_type(8)))  _Float16 v8h;
typedef __attribute__((ext_vector_type(8)))  float    v8f;

#define BATCH 8
#define LEN   2048
#define DIM   1024
#define SSEG  2048              // S_max = L
#define APAD  1032              // LDS row stride in halves (1024 + 8 pad)
#define MROWS 64                // rows of x staged per block (4 WMMA m-tiles)

// ---------------------------------------------------------------------------
// Kernel 0: W1 [k][n] f32 -> w1p, packed in WMMA B-fragment order:
//   w1p[((nt*32 + kt)*32 + lane)*16 + i] = W1[k][n]
//   with n = nt*16 + (lane&15), k = kt*32 + (lane>>4)*8 + (i&7) + (i>>3)*16
// so a wave's B-fragment load at (nt,kt) is one contiguous coalesced 1KB read.
// ---------------------------------------------------------------------------
__global__ __launch_bounds__(256) void convW1_kernel(const float* __restrict__ W1,
                                                     _Float16* __restrict__ w1p) {
  int idx = blockIdx.x * 256 + threadIdx.x;           // over 1024*1024
  int i    = idx & 15;
  int lane = (idx >> 4) & 31;
  int kt   = (idx >> 9) & 31;
  int nt   = idx >> 14;
  int n = nt * 16 + (lane & 15);
  int k = kt * 32 + (lane >> 4) * 8 + (i & 7) + (i >> 3) * 16;
  w1p[idx] = (_Float16)W1[(size_t)k * DIM + n];
}

// ---------------------------------------------------------------------------
// Kernel 1: fused  hard = [ relu(x@W1+b1)@W2 + b2 + gumbel(noise) > 0 ]
// 64-row strip per block (8 waves). Each wave: 8 n-tiles (processed in pairs),
// reusing each B fragment across 4 m-tiles => 8 WMMAs per B-fragment pair.
// ---------------------------------------------------------------------------
__global__ __launch_bounds__(256) void gemm_hard_kernel(
    const float* __restrict__ x, const float* __restrict__ noise,
    const _Float16* __restrict__ w1p, const float* __restrict__ b1v,
    const float* __restrict__ W2, const float* __restrict__ b2,
    float* __restrict__ hard)
{
  __shared__ _Float16 lA[MROWS * APAD];   // 64-row x strip, f16, padded (132 KB)
  __shared__ float lLogit[MROWS];

  const int tid = threadIdx.x;
  const int m0  = blockIdx.x * MROWS;     // strip start row in [0, B*L)

  // stage x strip -> LDS (f32 -> f16), coalesced float4 reads
  {
    const float4* x4 = (const float4*)(x + (size_t)m0 * DIM);
    for (int i = tid; i < MROWS * (DIM / 4); i += 256) {
      int r  = i >> 8;              // DIM/4 = 256 float4 per row
      int c4 = i & 255;
      float4 v = x4[(size_t)r * 256 + c4];
      _Float16* dst = &lA[r * APAD + c4 * 4];
      dst[0] = (_Float16)v.x; dst[1] = (_Float16)v.y;
      dst[2] = (_Float16)v.z; dst[3] = (_Float16)v.w;
    }
  }
  if (tid < MROWS) lLogit[tid] = 0.f;
  __syncthreads();

  const int wave = tid >> 5, lane = tid & 31;
  const int halfsel = lane >> 4;       // 0: lanes 0-15, 1: lanes 16-31
  const int lid = lane & 15;

  float pl[4][8];
  #pragma unroll
  for (int mt = 0; mt < 4; ++mt)
    #pragma unroll
    for (int j = 0; j < 8; ++j) pl[mt][j] = 0.f;

  // each wave owns n-tiles [wave*8, wave*8+8), processed in pairs
  #pragma unroll 1
  for (int p = 0; p < 4; ++p) {
    const int nt0 = wave * 8 + p * 2;
    v8f c[2][4];
    #pragma unroll
    for (int q = 0; q < 2; ++q)
      #pragma unroll
      for (int mt = 0; mt < 4; ++mt) c[q][mt] = (v8f){};

    for (int kt = 0; kt < 32; ++kt) {
      // coalesced packed B fragments for n-tiles nt0, nt0+1
      v16h bf0 = *(const v16h*)(w1p + ((size_t)(((nt0    ) * 32 + kt) * 32 + lane) << 4));
      v16h bf1 = *(const v16h*)(w1p + ((size_t)(((nt0 + 1) * 32 + kt) * 32 + lane) << 4));
      #pragma unroll
      for (int mt = 0; mt < 4; ++mt) {
        // A fragment (16x32 f16 layout): halves 0-7 = K+koff, 8-15 = K+koff+16
        const int ao = (mt * 16 + lid) * APAD + halfsel * 8 + kt * 32;
        v8h alo = *(const v8h*)(&lA[ao]);
        v8h ahi = *(const v8h*)(&lA[ao + 16]);
        v16h a;
        #pragma unroll
        for (int i = 0; i < 8; ++i) { a[i] = alo[i]; a[i + 8] = ahi[i]; }
        c[0][mt] = __builtin_amdgcn_wmma_f32_16x16x32_f16(
                       false, a, false, bf0, (short)0, c[0][mt], false, false);
        c[1][mt] = __builtin_amdgcn_wmma_f32_16x16x32_f16(
                       false, a, false, bf1, (short)0, c[1][mt], false, false);
      }
    }
    // fold this n-tile pair into per-row partial logits
    #pragma unroll
    for (int q = 0; q < 2; ++q) {
      const int ncol = (nt0 + q) * 16 + lid;
      const float w2n = W2[ncol];
      const float b1n = b1v[ncol];
      #pragma unroll
      for (int mt = 0; mt < 4; ++mt)
        #pragma unroll
        for (int j = 0; j < 8; ++j) {
          float hv = c[q][mt][j] + b1n;
          hv = hv > 0.f ? hv : 0.f;          // relu
          pl[mt][j] += hv * w2n;
        }
    }
  }

  // reduce partial logits across the 16 lanes of each half-group
  #pragma unroll
  for (int mt = 0; mt < 4; ++mt)
    #pragma unroll
    for (int j = 0; j < 8; ++j) {
      float v = pl[mt][j];
      v += __shfl_xor(v, 1, 16);
      v += __shfl_xor(v, 2, 16);
      v += __shfl_xor(v, 4, 16);
      v += __shfl_xor(v, 8, 16);
      if (lid == 0)
        atomicAdd(&lLogit[mt * 16 + halfsel * 8 + j], v);   // ds_add_f32
    }
  __syncthreads();

  if (tid < MROWS) {
    const int row = m0 + tid;
    float logit = lLogit[tid] + b2[0];
    float u = noise[row];
    u = fminf(fmaxf(u, 1e-6f), 1.f - 1e-6f);
    float g = logit + logf(u) - log1pf(-u);    // sigmoid(g) > 0.5  <=>  g > 0
    hard[row] = (g > 0.f) ? 1.f : 0.f;
  }
}

// ---------------------------------------------------------------------------
// Kernel 2: per-batch inclusive scan of hard -> seg_incl, total -> kcount
// ---------------------------------------------------------------------------
__global__ __launch_bounds__(256) void scan_kernel(const float* __restrict__ hard,
                                                   int* __restrict__ seg_incl,
                                                   int* __restrict__ kcount) {
  __shared__ int tot[256];
  const int b = blockIdx.x, t = threadIdx.x;
  const float* h = hard + (size_t)b * LEN;
  const int base = t * 8;
  int loc[8], run = 0;
  #pragma unroll
  for (int i = 0; i < 8; ++i) { run += (int)h[base + i]; loc[i] = run; }
  tot[t] = run;
  __syncthreads();
  for (int off = 1; off < 256; off <<= 1) {        // Hillis-Steele
    int v = (t >= off) ? tot[t - off] : 0;
    __syncthreads();
    tot[t] += v;
    __syncthreads();
  }
  const int excl = tot[t] - run;
  int* o = seg_incl + (size_t)b * LEN;
  #pragma unroll
  for (int i = 0; i < 8; ++i) o[base + i] = excl + loc[i];
  if (t == 255) kcount[b] = tot[255];
}

// ---------------------------------------------------------------------------
// Kernel 3a/3b: segment metadata (counts + first-token index per segment)
// ---------------------------------------------------------------------------
__global__ __launch_bounds__(256) void zero_counts_kernel(int* __restrict__ counts) {
  counts[blockIdx.x * 256 + threadIdx.x] = 0;
}

__global__ __launch_bounds__(256) void segmeta_kernel(const float* __restrict__ hard,
                                                      const int* __restrict__ seg_incl,
                                                      int* __restrict__ counts,
                                                      int* __restrict__ starts) {
  const int idx = blockIdx.x * 256 + threadIdx.x;   // 0 .. B*L-1
  const int b = idx >> 11, l = idx & (LEN - 1);
  const float* h = hard + (size_t)b * LEN;
  const int* si = seg_incl + (size_t)b * LEN;
  const int e = si[l] - (int)h[l];                  // exclusive cumsum = segment id
  atomicAdd(&counts[b * SSEG + e], 1);
  const int ep = (l == 0) ? -1 : si[l - 1] - (int)h[l - 1];
  if (e != ep) starts[b * SSEG + e] = l;            // unique writer per (b,e)
}

// ---------------------------------------------------------------------------
// Kernel 4: mean-pool each contiguous segment -> shortened [B, S+1, D]
// ---------------------------------------------------------------------------
__global__ __launch_bounds__(256) void pool_kernel(
    const float* __restrict__ x, const float* __restrict__ null_group,
    const int* __restrict__ counts, const int* __restrict__ starts,
    float* __restrict__ shortened) {
  const int blk = blockIdx.x;               // b*(S+1) + s'
  const int b = blk / (SSEG + 1);
  const int sp = blk % (SSEG + 1);
  const int d = threadIdx.x * 4;
  float* dst = shortened + (size_t)blk * DIM + d;
  if (sp == 0) {
    *(float4*)dst = *(const float4*)(null_group + d);
    return;
  }
  const int s = sp - 1;
  const int cnt = counts[b * SSEG + s];
  float4 acc = {0.f, 0.f, 0.f, 0.f};
  if (cnt > 0) {
    const int st = starts[b * SSEG + s];
    const float* xb = x + ((size_t)b * LEN + st) * DIM + d;
    for (int i = 0; i < cnt; ++i) {
      float4 v = *(const float4*)(xb + (size_t)i * DIM);
      acc.x += v.x; acc.y += v.y; acc.z += v.z; acc.w += v.w;
    }
    const float inv = 1.f / (float)cnt;
    acc.x *= inv; acc.y *= inv; acc.z *= inv; acc.w *= inv;
  }
  *(float4*)dst = acc;
}

// ---------------------------------------------------------------------------
// Kernel 5: upsample gather  upsampled[b,l,:] = shortened[b, seg_incl[b,l], :]
// ---------------------------------------------------------------------------
__global__ __launch_bounds__(256) void upsample_kernel(
    const int* __restrict__ seg_incl, const float* __restrict__ shortened,
    float* __restrict__ upsampled) {
  const int blk = blockIdx.x;               // b*L + l
  const int b = blk >> 11;
  const int d = threadIdx.x * 4;
  const int seg = seg_incl[blk];
  float4 v = *(const float4*)(shortened + ((size_t)b * (SSEG + 1) + seg) * DIM + d);
  *(float4*)(upsampled + (size_t)blk * DIM + d) = v;
}

// ---------------------------------------------------------------------------
// Kernel 6: binomial consistency loss (scalar)
// ---------------------------------------------------------------------------
__global__ void loss_kernel(const int* __restrict__ kcount, float* __restrict__ out_loss) {
  if (threadIdx.x == 0 && blockIdx.x == 0) {
    const float Lf = (float)LEN;
    float acc = 0.f;
    for (int b = 0; b < BATCH; ++b) {
      float k = (float)kcount[b];
      float logp = lgammaf(Lf + 1.f) - lgammaf(k + 1.f) - lgammaf(Lf - k + 1.f)
                 + k * logf(0.25f) + (Lf - k) * log1pf(-0.25f);
      acc += logp;
    }
    out_loss[0] = -(acc / (float)BATCH) / Lf;
  }
}

// ---------------------------------------------------------------------------
extern "C" void kernel_launch(void* const* d_in, const int* in_sizes, int n_in,
                              void* d_out, int out_size, void* d_ws, size_t ws_size,
                              hipStream_t stream) {
  (void)in_sizes; (void)n_in; (void)out_size; (void)ws_size;
  const float* x          = (const float*)d_in[0];   // [B,L,D]
  const float* noise      = (const float*)d_in[1];   // [B,L]
  const float* W1         = (const float*)d_in[2];   // [D,D]
  const float* b1         = (const float*)d_in[3];   // [D]
  const float* W2         = (const float*)d_in[4];   // [D,1]
  const float* b2         = (const float*)d_in[5];   // [1]
  const float* null_group = (const float*)d_in[6];   // [1,1,D]

  char* ws = (char*)d_ws;
  _Float16* w1p   = (_Float16*)ws;                                  // 2 MiB (packed)
  float*    hard  = (float*)(ws + (2u << 20));                      // 64 KiB
  int*      segi  = (int*)  (ws + (2u << 20) + (64u << 10));        // 64 KiB
  int*      cnts  = (int*)  (ws + (2u << 20) + (128u << 10));       // 64 KiB
  int*      strt  = (int*)  (ws + (2u << 20) + (192u << 10));       // 64 KiB
  int*      kcnt  = (int*)  (ws + (2u << 20) + (256u << 10));       // 32 B

  float* out_short = (float*)d_out;                                 // [B, S+1, D]
  float* out_up    = out_short + (size_t)BATCH * (SSEG + 1) * DIM;  // [B, L, D]
  float* out_loss  = out_up    + (size_t)BATCH * LEN * DIM;         // scalar

  hipLaunchKernelGGL(convW1_kernel, dim3((DIM * DIM) / 256), dim3(256), 0, stream, W1, w1p);
  hipLaunchKernelGGL(gemm_hard_kernel, dim3((BATCH * LEN) / MROWS), dim3(256), 0, stream,
                     x, noise, w1p, b1, W2, b2, hard);
  hipLaunchKernelGGL(scan_kernel, dim3(BATCH), dim3(256), 0, stream, hard, segi, kcnt);
  hipLaunchKernelGGL(zero_counts_kernel, dim3((BATCH * SSEG) / 256), dim3(256), 0, stream, cnts);
  hipLaunchKernelGGL(segmeta_kernel, dim3((BATCH * LEN) / 256), dim3(256), 0, stream,
                     hard, segi, cnts, strt);
  hipLaunchKernelGGL(pool_kernel, dim3(BATCH * (SSEG + 1)), dim3(256), 0, stream,
                     x, null_group, cnts, strt, out_short);
  hipLaunchKernelGGL(upsample_kernel, dim3(BATCH * LEN), dim3(256), 0, stream,
                     segi, out_short, out_up);
  hipLaunchKernelGGL(loss_kernel, dim3(1), dim3(1), 0, stream, kcnt, out_loss);
}
